// StackedLSTM_scratch_18597208391781
// MI455X (gfx1250) — matrix operand order repaired and checked by reading, not compiled, not run on hardware
//
#include <hip/hip_runtime.h>
#include <hip/hip_bf16.h>
#include <stdint.h>

// ---------------------------------------------------------------------------
// Types
// ---------------------------------------------------------------------------
typedef __bf16 bf16;
typedef __attribute__((ext_vector_type(16))) __bf16 v16bf;
typedef __attribute__((ext_vector_type(8)))  float  v8f;
typedef uint32_t u32;
typedef uint64_t u64;
typedef u32 u32x4 __attribute__((ext_vector_type(4)));
typedef u32 u32x8 __attribute__((ext_vector_type(8)));

#define BATCH    8
#define TSTEPS   512
#define HIDDEN   1024
#define GATES4   4096
#define NVOCAB   32000
#define NWG_LSTM 128   // 64 WGs own layer-0 cell tiles, 64 own layer-1 tiles

// ---------------------------------------------------------------------------
// Workspace layout (bytes)
// ---------------------------------------------------------------------------
static constexpr size_t OFF_BAR  = 0;                                   // barrier counter
static constexpr size_t OFF_H0B  = 256;                                 // h0 bf16 [2 parity][16][1024]
static constexpr size_t OFF_H1B  = OFF_H0B + 2ull*16*HIDDEN*2;
static constexpr size_t OFF_H0F  = OFF_H1B + 2ull*16*HIDDEN*2;          // h0 f32 [8][1024]
static constexpr size_t OFF_H1F  = OFF_H0F + 8ull*HIDDEN*4;
static constexpr size_t OFF_C0   = OFF_H1F + 8ull*HIDDEN*4;             // c f32 [8][1024]
static constexpr size_t OFF_C1   = OFF_C0  + 8ull*HIDDEN*4;
static constexpr size_t STATE_BYTES = OFF_C1 + 8ull*HIDDEN*4;

static constexpr size_t OFF_WXA  = ((STATE_BYTES + 255)/256)*256;       // wx bf16 [2][4096][1024]
static constexpr size_t OFF_WHA  = OFF_WXA + 2ull*GATES4*HIDDEN*2;      // wh bf16 [2][4096][1024]
static constexpr size_t OFF_EMBB = OFF_WHA + 2ull*GATES4*HIDDEN*2;      // emb bf16 [32000][1024]
static constexpr size_t OFF_XE   = OFF_EMBB + (size_t)NVOCAB*HIDDEN*2;  // xe bf16 [4096][1024], m=t*8+b
static constexpr size_t OFF_G0   = OFF_XE + 4096ull*HIDDEN*2;           // G0 f32 [512][4096][8]
static constexpr size_t OFF_OUT  = OFF_G0 + (size_t)TSTEPS*GATES4*8*4 + 256; // out bf16 [4096][1024], r=b*512+t

// ---------------------------------------------------------------------------
// Small helpers
// ---------------------------------------------------------------------------
__device__ __forceinline__ v8f vzero8() {
  v8f v;
#pragma unroll
  for (int i = 0; i < 8; ++i) v[i] = 0.0f;
  return v;
}

__device__ __forceinline__ v8f wmma_bf16(v16bf a, v16bf b, v8f c) {
  // D = A(16x32 bf16) * B(32x16 bf16) + C(16x16 f32)
  return __builtin_amdgcn_wmma_f32_16x16x32_bf16(false, a, false, b, (short)0, c,
                                                 false, false);
}

// A fragment: row-major W[ld] tile at (row_base, kbase).  ISA 16-bit A layout:
// lane<16 holds K = kbase+{0..7, 16..23}; lane>=16 holds K = kbase+{8..15, 24..31}.
__device__ __forceinline__ v16bf load_a_frag(const bf16* __restrict__ W, int ld,
                                             int row_base, int kbase) {
  const int lane = threadIdx.x & 31;
  const int m  = lane & 15;
  const int kb = kbase + ((lane >> 4) << 3);
  union { uint4 u[2]; v16bf v; } cv;
  cv.u[0] = *(const uint4*)(W + (size_t)(row_base + m) * ld + kb);
  cv.u[1] = *(const uint4*)(W + (size_t)(row_base + m) * ld + kb + 16);
  return cv.v;
}

// B fragment: B[k][n] = S[n][k] (S row-major, ld).  ISA 16-bit B layout:
// lane<16: col n=lane, K = kbase+{0..7, 16..23}; lane>=16: col n=lane-16, K = kbase+{8..15, 24..31}.
__device__ __forceinline__ v16bf load_b_frag(const bf16* __restrict__ S, int ld,
                                             int kbase) {
  const int lane = threadIdx.x & 31;
  const int n  = lane & 15;
  const int kb = kbase + ((lane >> 4) << 3);
  union { uint4 u[2]; v16bf v; } cv;
  const uint4* p = (const uint4*)(S + (size_t)n * ld + kb);
  cv.u[0] = p[0];
  cv.u[1] = p[2];
  return cv.v;
}

__device__ __forceinline__ u32 lds_addr_of(const void* p) {
  // LDS aperture: flat addr[31:0] is the LDS byte address.
  return (u32)(uintptr_t)p;
}

// Tensor Data Mover: 3D tile load (x=k, y=row, z=gate) into LDS.
// Inline asm (portable across ROCm7.2 / amdgpu-toolchain builtin arity skew).
__device__ __forceinline__ void tdm_load_3d(u32 lds_addr, u64 gaddr,
                                            u32 tdim0, u32 tdim1, u32 tdim2,
                                            u32 tile0, u32 tile1, u32 tile2,
                                            u64 str0, u64 str1, u64 str2) {
  u32x4 g0; u32x8 g1; u32x4 g2; u32x4 g3;
  // group0: count=1, lds_addr[63:32], global_addr[120:64], type=2
  g0[0] = 1u;
  g0[1] = lds_addr;
  g0[2] = (u32)gaddr;
  g0[3] = ((u32)(gaddr >> 32) & 0x01ffffffu) | (2u << 30);
  // group1: data_size=1 (2B), dims/tiles/strides
  g1[0] = 1u << 16;
  g1[1] = (tdim0 & 0xffffu) << 16;
  g1[2] = (tdim0 >> 16) | ((tdim1 & 0xffffu) << 16);
  g1[3] = (tdim1 >> 16) | (tile0 << 16);
  g1[4] = (tile1 & 0xffffu) | (tile2 << 16);
  g1[5] = (u32)str0;
  g1[6] = ((u32)(str0 >> 32) & 0xffffu) | (((u32)str1 & 0xffffu) << 16);
  g1[7] = (u32)(str1 >> 16);
  // group2: tensor_dim2, tensor_dim3=1, tensor_dim2_stride, tile_dim3=0
  g2[0] = tdim2;
  g2[1] = 1u;
  g2[2] = (u32)str2;
  g2[3] = (u32)(str2 >> 32) & 0xffffu;
  // group3: dim3 stride 0, tensor_dim4=1, tile_dim4=0
  g3[0] = 0u; g3[1] = 1u << 16; g3[2] = 0u; g3[3] = 0u;
  asm volatile("tensor_load_to_lds %0, %1, %2, %3"
               :: "s"(g0), "s"(g1), "s"(g2), "s"(g3)
               : "memory");
}

__device__ __forceinline__ void grid_barrier(u32* cnt) {
  __threadfence();
  if (threadIdx.x == 0) {
    u32 arr = atomicAdd(cnt, 1u) + 1u;
    u32 target = ((arr + NWG_LSTM - 1u) / NWG_LSTM) * NWG_LSTM;
    while (__hip_atomic_load(cnt, __ATOMIC_ACQUIRE, __HIP_MEMORY_SCOPE_AGENT) < target) {
      __builtin_amdgcn_s_sleep(1);
    }
  }
  __syncthreads();
  __threadfence();
}

// ---------------------------------------------------------------------------
// K1: f32 -> bf16 conversion (weights / embedding to L2-resident bf16)
// ---------------------------------------------------------------------------
__global__ void cvt_f32_bf16_kernel(const float* __restrict__ src,
                                    bf16* __restrict__ dst, size_t n) {
  size_t i = (size_t)blockIdx.x * blockDim.x + threadIdx.x;
  size_t stride = (size_t)gridDim.x * blockDim.x;
  for (; i < n; i += stride) dst[i] = (bf16)src[i];
}

// ---------------------------------------------------------------------------
// K2: embedding gather -> xe bf16 [m=t*8+b][k]
// ---------------------------------------------------------------------------
__global__ void embed_kernel(const long long* __restrict__ x,
                             const float* __restrict__ emb,
                             bf16* __restrict__ xe) {
  size_t i = (size_t)blockIdx.x * blockDim.x + threadIdx.x;
  size_t stride = (size_t)gridDim.x * blockDim.x;
  const size_t n = (size_t)TSTEPS * BATCH * HIDDEN;
  for (; i < n; i += stride) {
    int k = (int)(i & (HIDDEN - 1));
    int m = (int)(i >> 10);
    int b = m & 7, t = m >> 3;
    long long tok = x[b * TSTEPS + t];
    xe[i] = (bf16)emb[(size_t)tok * HIDDEN + k];
  }
}

// ---------------------------------------------------------------------------
// K3: G0[t][j][b] = (xe @ wx0^T)[m=t*8+b][j] + b0[j]   (WMMA, A=wx0, B=xe^T)
// ---------------------------------------------------------------------------
__global__ void __launch_bounds__(32)
gemm_g0_kernel(const bf16* __restrict__ wx0,   // [4096][1024]
               const bf16* __restrict__ xe,    // [4096][1024]
               const float* __restrict__ b0,   // [4096]
               float* __restrict__ G0) {       // [512][4096][8]
  const int jbase = blockIdx.x * 16;
  const int mbase = blockIdx.y * 16;
  const int lane  = threadIdx.x & 31;
  v8f acc = vzero8();
  for (int k = 0; k < HIDDEN; k += 32) {
    v16bf a = load_a_frag(wx0, HIDDEN, jbase, k);
    v16bf b = load_b_frag(xe + (size_t)mbase * HIDDEN, HIDDEN, k);
    acc = wmma_bf16(a, b, acc);
  }
  const int m  = mbase + (lane & 15);
  const int tt = m >> 3, bb = m & 7;
  const int jr = jbase + ((lane >> 4) << 3);
#pragma unroll
  for (int i = 0; i < 8; ++i) {
    G0[((size_t)tt * GATES4 + jr + i) * 8 + bb] = acc[i] + b0[jr + i];
  }
}

// ---------------------------------------------------------------------------
// K4: persistent sequential LSTM.  One wave per WG; WGs 0..63 = layer0 tiles,
//     WGs 64..127 = layer1 tiles.  Weight slices staged into LDS by the TDM,
//     double-buffered; h state double-buffered by timestep parity.
// ---------------------------------------------------------------------------
__device__ __forceinline__ void accum_gates(v8f* acc,                 // [4]
                                            const bf16* __restrict__ W, // [4096][1024]
                                            int cell0,
                                            const bf16* __restrict__ hsrc, // [16][1024]
                                            bf16* stage) {            // LDS [2][4*16*128]
  const u64 wbase = (u64)(uintptr_t)W;
  const u64 str1  = (u64)HIDDEN * (u64)HIDDEN;  // gate block stride (elements)
  // prologue: chunk 0
  tdm_load_3d(lds_addr_of(stage),
              wbase + 2ull * ((u64)cell0 * HIDDEN),
              HIDDEN, GATES4, 4, 128, 16, 4, HIDDEN, str1, 0);
  for (int kc = 0; kc < 8; ++kc) {
    if (kc + 1 < 8) {
      // make sure pending LDS reads of the buffer we're about to refill are done
      asm volatile("s_wait_dscnt 0x0" ::: "memory");
      tdm_load_3d(lds_addr_of(stage + ((kc + 1) & 1) * (4 * 16 * 128)),
                  wbase + 2ull * ((u64)cell0 * HIDDEN + (u64)(kc + 1) * 128),
                  HIDDEN, GATES4, 4, 128, 16, 4, HIDDEN, str1, 0);
      __builtin_amdgcn_s_wait_tensorcnt(1);   // chunk kc landed (in-order)
    } else {
      __builtin_amdgcn_s_wait_tensorcnt(0);
    }
    const bf16* buf = stage + (kc & 1) * (4 * 16 * 128);
    for (int kw = 0; kw < 4; ++kw) {
      v16bf bfrag = load_b_frag(hsrc, HIDDEN, kc * 128 + kw * 32);
#pragma unroll
      for (int gt = 0; gt < 4; ++gt) {
        v16bf afrag = load_a_frag(buf + gt * (16 * 128), 128, 0, kw * 32);
        acc[gt] = wmma_bf16(afrag, bfrag, acc[gt]);
      }
    }
  }
}

__global__ void __launch_bounds__(32)
lstm_seq_kernel(const bf16* __restrict__ whA,   // [2][4096][1024]
                const bf16* __restrict__ wxA,   // [2][4096][1024]
                const float* __restrict__ G0,   // [512][4096][8]
                const float* __restrict__ b1,   // wx_b layer1 [4096]
                bf16* __restrict__ h0b, bf16* __restrict__ h1b,   // [2][16][1024]
                float* __restrict__ h0f, float* __restrict__ h1f, // [8][1024]
                float* __restrict__ c0s, float* __restrict__ c1s, // [8][1024]
                bf16* __restrict__ outb,        // [4096][1024], row r=b*512+t
                u32* __restrict__ bar) {
  __shared__ __align__(16) bf16 stage[2][4 * 16 * 128];   // 32 KB double buffer

  const int g      = blockIdx.x;
  const int lane   = threadIdx.x & 31;
  const bool is_l1 = (g >= 64);
  const int cell0  = (is_l1 ? g - 64 : g) * 16;
  const int n      = lane & 15;            // D column = batch
  const int moff   = (lane >> 4) << 3;     // D row offset (cells)
  const bool bval  = (n < BATCH);

  const bf16* WH0 = whA;
  const bf16* WH1 = whA + (size_t)GATES4 * HIDDEN;
  const bf16* WX1 = wxA + (size_t)GATES4 * HIDDEN;

  for (int t = 0; t < TSTEPS; ++t) {
    const int p = t & 1;
    // ---------------- phase A: layer 0 ----------------
    if (!is_l1) {
      const bf16* hprev = h0b + (size_t)p * 16 * HIDDEN;
      v8f acc[4] = { vzero8(), vzero8(), vzero8(), vzero8() };
      accum_gates(acc, WH0, cell0, hprev, &stage[0][0]);
      if (bval) {
        const int b = n;
        const int cellb = cell0 + moff;
#pragma unroll
        for (int i = 0; i < 8; ++i) {
          const int cell = cellb + i;
          float gi = acc[0][i] + G0[((size_t)t * GATES4 + cell       ) * 8 + b];
          float gf = acc[1][i] + G0[((size_t)t * GATES4 + cell + 1024) * 8 + b];
          float gg = acc[2][i] + G0[((size_t)t * GATES4 + cell + 2048) * 8 + b];
          float go = acc[3][i] + G0[((size_t)t * GATES4 + cell + 3072) * 8 + b];
          float co = c0s[b * HIDDEN + cell];
          float si = 1.f / (1.f + __expf(-gi));
          float sf = 1.f / (1.f + __expf(-gf));
          float so = 1.f / (1.f + __expf(-go));
          float cn = sf * co + si * tanhf(gg);
          float hn = so * tanhf(co);              // repo quirk: tanh of OLD cell
          c0s[b * HIDDEN + cell] = cn;
          h0f[b * HIDDEN + cell] = hn;
          h0b[(size_t)(1 - p) * 16 * HIDDEN + b * HIDDEN + cell] = (bf16)hn;
        }
      }
    }
    grid_barrier(bar);
    // ---------------- phase B: layer 1 ----------------
    if (is_l1) {
      const bf16* h0new  = h0b + (size_t)(1 - p) * 16 * HIDDEN;
      const bf16* h1prev = h1b + (size_t)p * 16 * HIDDEN;
      v8f acc[4] = { vzero8(), vzero8(), vzero8(), vzero8() };
      accum_gates(acc, WX1, cell0, h0new,  &stage[0][0]);
      accum_gates(acc, WH1, cell0, h1prev, &stage[0][0]);
      if (bval) {
        const int b = n;
        const int cellb = cell0 + moff;
#pragma unroll
        for (int i = 0; i < 8; ++i) {
          const int cell = cellb + i;
          float gi = acc[0][i] + b1[cell];
          float gf = acc[1][i] + b1[cell + 1024];
          float gg = acc[2][i] + b1[cell + 2048];
          float go = acc[3][i] + b1[cell + 3072];
          float co = c1s[b * HIDDEN + cell];
          float si = 1.f / (1.f + __expf(-gi));
          float sf = 1.f / (1.f + __expf(-gf));
          float so = 1.f / (1.f + __expf(-go));
          float cn = sf * co + si * tanhf(gg);
          float hn = so * tanhf(co);
          c1s[b * HIDDEN + cell] = cn;
          h1f[b * HIDDEN + cell] = hn;
          h1b[(size_t)(1 - p) * 16 * HIDDEN + b * HIDDEN + cell] = (bf16)hn;
          outb[((size_t)b * TSTEPS + t) * HIDDEN + cell] = (bf16)hn;
        }
      }
    }
    grid_barrier(bar);
  }
}

// ---------------------------------------------------------------------------
// K5: logits[r][v] = out[r][:] . emb[v][:]   (WMMA, A=out rows, B=emb^T)
//     r = b*512+t, so logits rows are already in [B,T,V] order.
// ---------------------------------------------------------------------------
__global__ void __launch_bounds__(32)
gemm_logits_kernel(const bf16* __restrict__ outb,  // [4096][1024]
                   const bf16* __restrict__ embB,  // [32000][1024]
                   float* __restrict__ logits) {   // [4096][32000]
  const int vt4   = blockIdx.x;        // 4 v-tiles per wave (A reuse)
  const int rbase = blockIdx.y * 16;
  const int lane  = threadIdx.x & 31;
  v8f acc[4] = { vzero8(), vzero8(), vzero8(), vzero8() };
  for (int k = 0; k < HIDDEN; k += 32) {
    v16bf a = load_a_frag(outb, HIDDEN, rbase, k);
#pragma unroll
    for (int q = 0; q < 4; ++q) {
      const int vbase = (vt4 * 4 + q) * 16;
      v16bf b = load_b_frag(embB + (size_t)vbase * HIDDEN, HIDDEN, k);
      acc[q] = wmma_bf16(a, b, acc[q]);
    }
  }
  const int r = rbase + ((lane >> 4) << 3);
#pragma unroll
  for (int q = 0; q < 4; ++q) {
    const int v = (vt4 * 4 + q) * 16 + (lane & 15);
#pragma unroll
    for (int i = 0; i < 8; ++i) {
      logits[(size_t)(r + i) * NVOCAB + v] = acc[q][i];
    }
  }
}

// ---------------------------------------------------------------------------
// K6: copy final h/c into d_out tail: [2][8][1024] h then [2][8][1024] c
// ---------------------------------------------------------------------------
__global__ void finalize_kernel(const float* __restrict__ h0f,
                                const float* __restrict__ h1f,
                                const float* __restrict__ c0s,
                                const float* __restrict__ c1s,
                                float* __restrict__ dst) {
  int i = blockIdx.x * blockDim.x + threadIdx.x;
  if (i < BATCH * HIDDEN) {
    dst[i]                       = h0f[i];
    dst[BATCH * HIDDEN + i]      = h1f[i];
    dst[2 * BATCH * HIDDEN + i]  = c0s[i];
    dst[3 * BATCH * HIDDEN + i]  = c1s[i];
  }
}

// ---------------------------------------------------------------------------
// Launch
// ---------------------------------------------------------------------------
extern "C" void kernel_launch(void* const* d_in, const int* in_sizes, int n_in,
                              void* d_out, int out_size, void* d_ws, size_t ws_size,
                              hipStream_t stream) {
  (void)in_sizes; (void)n_in; (void)out_size; (void)ws_size;
  const long long* x   = (const long long*)d_in[0];   // int64 tokens [8][512]
  const float* emb     = (const float*)d_in[1];       // [32000][1024]
  const float* wx_w    = (const float*)d_in[2];       // [2][4096][1024]
  const float* wx_b    = (const float*)d_in[3];       // [2][4096]
  const float* wh_w    = (const float*)d_in[4];       // [2][4096][1024]
  char* ws = (char*)d_ws;

  u32*   bar  = (u32*)  (ws + OFF_BAR);
  bf16*  h0b  = (bf16*) (ws + OFF_H0B);
  bf16*  h1b  = (bf16*) (ws + OFF_H1B);
  float* h0f  = (float*)(ws + OFF_H0F);
  float* h1f  = (float*)(ws + OFF_H1F);
  float* c0s  = (float*)(ws + OFF_C0);
  float* c1s  = (float*)(ws + OFF_C1);
  bf16*  wxA  = (bf16*) (ws + OFF_WXA);
  bf16*  whA  = (bf16*) (ws + OFF_WHA);
  bf16*  embB = (bf16*) (ws + OFF_EMBB);
  bf16*  xe   = (bf16*) (ws + OFF_XE);
  float* G0   = (float*)(ws + OFF_G0);
  bf16*  outb = (bf16*) (ws + OFF_OUT);
  float* logits = (float*)d_out;
  float* hc_out = logits + (size_t)BATCH * TSTEPS * NVOCAB;

  // zero barrier + h/c state every call (deterministic, capture-legal)
  hipMemsetAsync(ws, 0, STATE_BYTES, stream);

  // bf16 conversions
  cvt_f32_bf16_kernel<<<2048, 256, 0, stream>>>(wx_w, wxA, (size_t)2 * GATES4 * HIDDEN);
  cvt_f32_bf16_kernel<<<2048, 256, 0, stream>>>(wh_w, whA, (size_t)2 * GATES4 * HIDDEN);
  cvt_f32_bf16_kernel<<<4096, 256, 0, stream>>>(emb, embB, (size_t)NVOCAB * HIDDEN);

  // embedding gather
  embed_kernel<<<2048, 256, 0, stream>>>(x, emb, xe);

  // hoisted layer-0 input projection: G0 = xe @ wx0^T + b0
  gemm_g0_kernel<<<dim3(GATES4 / 16, (TSTEPS * BATCH) / 16), 32, 0, stream>>>(
      wxA, xe, wx_b, G0);

  // persistent sequential LSTM (TDM-staged weights, grid-barrier per phase)
  lstm_seq_kernel<<<NWG_LSTM, 32, 0, stream>>>(
      whA, wxA, G0, wx_b + GATES4, h0b, h1b, h0f, h1f, c0s, c1s, outb, bar);

  // tied decoder: logits = out @ emb^T
  gemm_logits_kernel<<<dim3(NVOCAB / 64, (TSTEPS * BATCH) / 16), 32, 0, stream>>>(
      outb, embB, logits);

  // final h, c
  finalize_kernel<<<(BATCH * HIDDEN + 255) / 256, 256, 0, stream>>>(
      h0f, h1f, c0s, c1s, hc_out);
}
